// AttentiveAtlasEncoder_89215060673150
// MI455X (gfx1250) — matrix operand order repaired and checked by reading, not compiled, not run on hardware
//
#include <hip/hip_runtime.h>
#include <hip/hip_bf16.h>
#include <math.h>

// ---------------------------------------------------------------------------
// CDNA5 / gfx1250 wave32 WMMA types
// ---------------------------------------------------------------------------
typedef __attribute__((ext_vector_type(16))) __bf16 v16bf;
typedef __attribute__((ext_vector_type(8)))  float  v8f;
typedef __attribute__((ext_vector_type(4)))  unsigned u32x4;
typedef __attribute__((ext_vector_type(8)))  int    i32x8;
typedef __attribute__((ext_vector_type(4)))  int    i32x4;

#define WMMA_BF16(a, b, c) \
  __builtin_amdgcn_wmma_f32_16x16x32_bf16(false, (a), false, (b), (short)0, (c), false, false)

// Branch-free exact-ish GELU (tanh form): one v_exp_f32 + rcp instead of the
// branchy libm erff expansion that blew up the VALU count in round 1.
__device__ __forceinline__ float gelu_f(float x) {
  float x3 = x * x * x;
  float y = 0.7978845608028654f * (x + 0.044715f * x3);
  float e = __expf(2.0f * y);
  float t = 1.0f - 2.0f / (e + 1.0f);
  return 0.5f * x * (1.0f + t);
}

// ---------------------------------------------------------------------------
// TDM availability (ROCm 7.2 clang-22: 5-arg builtin; clang-23 toolchain: 6-arg)
// ---------------------------------------------------------------------------
#ifdef __HIP_DEVICE_COMPILE__
#if __has_builtin(__builtin_amdgcn_tensor_load_to_lds) && \
    __has_builtin(__builtin_amdgcn_s_wait_tensorcnt)
#define HAVE_TDM 1
#else
#define HAVE_TDM 0
#endif
#else
#define HAVE_TDM 0
#endif

// ---------------------------------------------------------------------------
// Fragment loaders (wave32; this file's contiguous k order, same for A and B).
// A (16x32, 16-bit): lanes 0-15 -> M=lane,    elems 0..15 = K 0..15
//                    lanes16-31 -> M=lane-16, elems 0..15 = K 16..31
// B (32x16, 16-bit): lanes 0-15 -> N=lane,   elems 0..15 = K 0..15
//                    lanes16-31 -> N=lane-16, elems 0..15 = K 16..31
// C/D (16x16 f32):   lane half selects M base (0 or 8), N = lane&15, 8 VGPRs.
// ---------------------------------------------------------------------------
__device__ __forceinline__ v16bf load_a_f32(const float* src, int ld, int row0, int k0) {
  int lane = threadIdx.x & 31;
  const float* p = src + (row0 + (lane & 15)) * ld + k0 + ((lane >> 4) & 1) * 16;
  v16bf a;
#pragma unroll
  for (int i = 0; i < 8; ++i) a[i] = (__bf16)p[i];
#pragma unroll
  for (int i = 0; i < 8; ++i) a[8 + i] = (__bf16)p[8 + i];
  return a;
}

__device__ __forceinline__ v16bf load_a_bf16(const __bf16* src, int ld, int row0, int k0) {
  int lane = threadIdx.x & 31;
  const __bf16* p = src + (row0 + (lane & 15)) * ld + k0 + ((lane >> 4) & 1) * 16;
  v16bf a;
#pragma unroll
  for (int i = 0; i < 8; ++i) a[i] = p[i];
#pragma unroll
  for (int i = 0; i < 8; ++i) a[8 + i] = p[8 + i];
  return a;
}

__device__ __forceinline__ v16bf load_a_f32_pad(const float* src, int ld, int row0, int kvalid) {
  int lane = threadIdx.x & 31;
  int m = row0 + (lane & 15);
  int kh = (lane >> 4) & 1;
  v16bf a;
#pragma unroll
  for (int i = 0; i < 8; ++i) {
    int k = kh * 16 + i;
    a[i] = (k < kvalid) ? (__bf16)src[m * ld + k] : (__bf16)0.0f;
  }
#pragma unroll
  for (int i = 0; i < 8; ++i) {
    int k = kh * 16 + 8 + i;
    a[8 + i] = (k < kvalid) ? (__bf16)src[m * ld + k] : (__bf16)0.0f;
  }
  return a;
}

// B fragment from row-major bf16 W[N][K] computing X @ W^T: 32 contiguous bytes/lane
__device__ __forceinline__ v16bf load_b_bf16(const __bf16* W, int ld, int n0, int k0) {
  int lane = threadIdx.x & 31;
  const __bf16* p = W + (n0 + (lane & 15)) * ld + k0 + ((lane >> 4) & 1) * 16;
  v16bf b;
#pragma unroll
  for (int i = 0; i < 16; ++i) b[i] = p[i];
  return b;
}

__device__ __forceinline__ v16bf load_b_f32(const float* W, int ld, int n0, int k0) {
  int lane = threadIdx.x & 31;
  const float* p = W + (n0 + (lane & 15)) * ld + k0 + ((lane >> 4) & 1) * 16;
  v16bf b;
#pragma unroll
  for (int i = 0; i < 16; ++i) b[i] = (__bf16)p[i];
  return b;
}

__device__ __forceinline__ v16bf load_b_f32_pad(const float* W, int ld, int n0, int kvalid) {
  int lane = threadIdx.x & 31;
  int n = n0 + (lane & 15);
  int kbase = ((lane >> 4) & 1) * 16;
  v16bf b;
#pragma unroll
  for (int i = 0; i < 16; ++i) {
    int k = kbase + i;
    b[i] = (k < kvalid) ? (__bf16)W[n * ld + k] : (__bf16)0.0f;
  }
  return b;
}

// B fragment from an LDS-staged [N][32] bf16 tile (k0 implicit 0, ld = 32)
__device__ __forceinline__ v16bf load_b_lds(const __bf16* wt, int n0) {
  int lane = threadIdx.x & 31;
  const __bf16* p = wt + (n0 + (lane & 15)) * 32 + ((lane >> 4) & 1) * 16;
  v16bf b;
#pragma unroll
  for (int i = 0; i < 16; ++i) b[i] = p[i];
  return b;
}

// ---------------------------------------------------------------------------
// Problem constants
// ---------------------------------------------------------------------------
#define BDIM   32768
#define IN_DIM 64
#define HID    256
#define LAT    32
#define NC     8
#define CODES  256

// output offsets (floats, concatenated tuple order)
#define OFF_KCHART 0
#define OFF_KCODE  32768
#define OFF_ZN     65536
#define OFF_ZTEX   1114112
#define OFF_RW     2162688
#define OFF_ZGEO   2424832
#define OFF_VQ     3473408
#define OFF_IDX    3473409
#define OFF_ZNALL  3735553

// ---------------------------------------------------------------------------
// TDM: async-load one [256 rows x 32 cols] bf16 tile of W2h into LDS.
// D# per ISA ch.8: group0 = {count, lds_addr, global_addr, type=2},
// group1 = {data_size=2B, tensor_dim0=256, tensor_dim1=256, tile_dim0=32,
//           tile_dim1=256, tensor_dim0_stride=256}.
// ---------------------------------------------------------------------------
#if HAVE_TDM
__device__ __forceinline__ void tdm_load_tile(unsigned lds_off, const void* gptr) {
  unsigned long long ga = (unsigned long long)(uintptr_t)gptr;
  u32x4 g0;
  g0[0] = 1u;                                           // count=1, user mode
  g0[1] = lds_off;                                      // lds_addr (bytes)
  g0[2] = (unsigned)(ga & 0xffffffffu);                 // global_addr[31:0]
  g0[3] = (unsigned)((ga >> 32) & 0x1ffffffu) | (2u << 30); // addr[56:32] | type=2
  i32x8 g1;
  g1[0] = (1 << 16);                 // workgroup_mask=0, data_size=1 (2 bytes)
  g1[1] = (int)(256u << 16);         // tensor_dim0[15:0] in bits[63:48]
  g1[2] = (int)(256u << 16);         // tensor_dim0[31:16]=0 | tensor_dim1[15:0]=256
  g1[3] = (int)(32u << 16);          // tensor_dim1[31:16]=0 | tile_dim0=32
  g1[4] = (int)(256u);               // tile_dim1=256, tile_dim2=0
  g1[5] = 256;                       // tensor_dim0_stride[31:0]
  g1[6] = 0;                         // stride[47:32]=0, tensor_dim1_stride lo=0
  g1[7] = 0;
  i32x4 z4 = {0, 0, 0, 0};
#if __clang_major__ >= 23
  i32x8 z8 = {0, 0, 0, 0, 0, 0, 0, 0};
  __builtin_amdgcn_tensor_load_to_lds(g0, g1, z4, z4, z8, 0);
#else
  __builtin_amdgcn_tensor_load_to_lds(g0, g1, z4, z4, 0);
#endif
}
#endif

// ---------------------------------------------------------------------------
// K0: fold router GEMM Mq = cq@Wk (rows 8..15 zero, bq pad = -1e30), codebook
//     norms, and one-time f32 -> bf16 conversion of all weight panels.
// ---------------------------------------------------------------------------
__global__ void k0_precompute(const float* __restrict__ Wk, const float* __restrict__ bk,
                              const float* __restrict__ cq, const float* __restrict__ cb,
                              const float* __restrict__ W1, const float* __restrict__ W2,
                              const float* __restrict__ Wv,
                              __bf16* __restrict__ W1h, __bf16* __restrict__ W2h,
                              __bf16* __restrict__ Mqh, __bf16* __restrict__ Wvh,
                              __bf16* __restrict__ cbh,
                              float* __restrict__ bq, float* __restrict__ csq) {
  int tid = threadIdx.x;
  for (int idx = tid; idx < 16 * HID; idx += 256) {
    int n = idx >> 8, e = idx & 255;
    float acc = 0.0f;
    if (n < NC) {
#pragma unroll 4
      for (int d = 0; d < HID; ++d) acc += cq[n * HID + d] * Wk[d * HID + e];
    }
    Mqh[idx] = (__bf16)acc;
  }
  if (tid < 16) {
    float acc = -1.0e30f;
    if (tid < NC) {
      acc = 0.0f;
      for (int d = 0; d < HID; ++d) acc += cq[tid * HID + d] * bk[d];
    }
    bq[tid] = acc;
  }
  for (int idx = tid; idx < NC * CODES; idx += 256) {
    const float* p = cb + idx * LAT;
    float acc = 0.0f;
#pragma unroll
    for (int d = 0; d < LAT; ++d) acc += p[d] * p[d];
    csq[idx] = acc;
  }
  for (int idx = tid; idx < HID * IN_DIM; idx += 256) W1h[idx] = (__bf16)W1[idx];
  for (int idx = tid; idx < HID * HID; idx += 256)    W2h[idx] = (__bf16)W2[idx];
  for (int idx = tid; idx < LAT * HID; idx += 256)    Wvh[idx] = (__bf16)Wv[idx];
  for (int idx = tid; idx < NC * CODES * LAT; idx += 256) cbh[idx] = (__bf16)cb[idx];
}

// ---------------------------------------------------------------------------
// K1: h1 = gelu(x @ W1^T + b1), bf16 out. 256 thr = 8 waves, 64 rows/block.
// ---------------------------------------------------------------------------
__global__ void __launch_bounds__(256) k1_gemm(const float* __restrict__ x,
                                               const __bf16* __restrict__ W1h,
                                               const float* __restrict__ b1,
                                               __bf16* __restrict__ h1) {
  __builtin_prefetch(W1h + (threadIdx.x << 6), 0, 1);  // global_prefetch_b8

  int w = threadIdx.x >> 5;
  int lane = threadIdx.x & 31;
  int row0 = blockIdx.x * 64 + (w & 3) * 16;
  int col = lane & 15;
  int mbase = (lane >> 4) * 8;

  v16bf a0 = load_a_f32(x, IN_DIM, row0, 0);
  v16bf a1 = load_a_f32(x, IN_DIM, row0, 32);

#pragma unroll
  for (int j = 0; j < 8; ++j) {
    int c0 = ((w >> 2) * 8 + j) * 16;
    float bias = b1[c0 + col];
    v8f acc;
#pragma unroll
    for (int r = 0; r < 8; ++r) acc[r] = bias;
    v16bf bf = load_b_bf16(W1h, IN_DIM, c0, 0);
    acc = WMMA_BF16(a0, bf, acc);
    bf = load_b_bf16(W1h, IN_DIM, c0, 32);
    acc = WMMA_BF16(a1, bf, acc);
#pragma unroll
    for (int r = 0; r < 8; ++r)
      h1[(row0 + mbase + r) * HID + c0 + col] = (__bf16)gelu_f(acc[r]);
  }
}

// ---------------------------------------------------------------------------
// K2: h = gelu(h1 @ W2^T + b2). W2 k-panels (256x32 bf16, 16 KB) are staged
// into LDS by the Tensor Data Mover, double-buffered: wave 0 issues
// tensor_load_to_lds for tile ks+1, waits s_wait_tensorcnt for tile ks, and a
// workgroup barrier publishes it to the other 7 waves.
// ---------------------------------------------------------------------------
__global__ void __launch_bounds__(256) k2_gemm(const __bf16* __restrict__ h1,
                                               const __bf16* __restrict__ W2h,
                                               const float* __restrict__ b2,
                                               __bf16* __restrict__ h) {
  __shared__ __bf16 wtile[2][HID * 32];  // 2 x 16 KB

  int tid = threadIdx.x, w = tid >> 5, lane = tid & 31;
  int row0 = blockIdx.x * 64 + (w & 3) * 16;
  int col = lane & 15, mbase = (lane >> 4) * 8;

  v16bf a[8];
#pragma unroll
  for (int ks = 0; ks < 8; ++ks) a[ks] = load_a_bf16(h1, HID, row0, ks * 32);

  v8f acc[8];
#pragma unroll
  for (int j = 0; j < 8; ++j) {
    float bias = b2[((w >> 2) * 8 + j) * 16 + col];
#pragma unroll
    for (int r = 0; r < 8; ++r) acc[j][r] = bias;
  }

#if HAVE_TDM
  if (w == 0) tdm_load_tile((unsigned)(uintptr_t)&wtile[0][0], W2h);
  for (int ks = 0; ks < 8; ++ks) {
    if (w == 0) {
      if (ks < 7) {
        tdm_load_tile((unsigned)(uintptr_t)&wtile[(ks + 1) & 1][0], W2h + (ks + 1) * 32);
        __builtin_amdgcn_s_wait_tensorcnt(1);  // tile ks done, ks+1 in flight
      } else {
        __builtin_amdgcn_s_wait_tensorcnt(0);  // last tile done
      }
    }
    __syncthreads();
    const __bf16* wt = &wtile[ks & 1][0];
#pragma unroll
    for (int j = 0; j < 8; ++j) {
      v16bf bf = load_b_lds(wt, ((w >> 2) * 8 + j) * 16);
      acc[j] = WMMA_BF16(a[ks], bf, acc[j]);
    }
    __syncthreads();
  }
#else
  for (int ks = 0; ks < 8; ++ks) {
    __syncthreads();
    {  // cooperative stage: thread t copies row t's 32 bf16 (16 dwords)
      const unsigned* src = (const unsigned*)W2h + tid * (HID / 2) + ks * 16;
      unsigned* dst = (unsigned*)&wtile[0][0] + tid * 16;
#pragma unroll
      for (int i = 0; i < 16; ++i) dst[i] = src[i];
    }
    __syncthreads();
    const __bf16* wt = &wtile[0][0];
#pragma unroll
    for (int j = 0; j < 8; ++j) {
      v16bf bf = load_b_lds(wt, ((w >> 2) * 8 + j) * 16);
      acc[j] = WMMA_BF16(a[ks], bf, acc[j]);
    }
  }
#endif

#pragma unroll
  for (int j = 0; j < 8; ++j) {
    int c0 = ((w >> 2) * 8 + j) * 16;
#pragma unroll
    for (int r = 0; r < 8; ++r)
      h[(row0 + mbase + r) * HID + c0 + col] = (__bf16)gelu_f(acc[j][r]);
  }
}

// ---------------------------------------------------------------------------
// K3: fused epilogue. 128 thr = 4 waves; each wave owns 16 rows end-to-end.
// ---------------------------------------------------------------------------
__global__ void __launch_bounds__(128) k3_fused(
    const __bf16* __restrict__ h, const __bf16* __restrict__ Mqh,
    const float* __restrict__ bq, const float* __restrict__ csq,
    const __bf16* __restrict__ Wvh, const float* __restrict__ bv,
    const float* __restrict__ cb, const __bf16* __restrict__ cbh,
    const float* __restrict__ Ws1, const float* __restrict__ bs1,
    const float* __restrict__ Ws2, const float* __restrict__ bs2,
    float* __restrict__ out, float* __restrict__ part) {
  __shared__ float vsh[4][512];
  __shared__ float deltash[4][512];
  __shared__ float zqbsh[4][512];
  __shared__ float znbsh[4][512];
  __shared__ float tsh[4][256];
  __shared__ float rwsh[4][256];
  __shared__ int   idxsh[4][128];
  __shared__ int   kchsh[4][16];
  __shared__ float redsh[4];

  int tid = threadIdx.x;
  int w = tid >> 5;
  int lane = tid & 31;
  int col = lane & 15;
  int mbase = (lane >> 4) * 8;
  int gr0 = blockIdx.x * 64 + w * 16;

  float* out_Kchart = out + OFF_KCHART;
  float* out_Kcode  = out + OFF_KCODE;
  float* out_zn     = out + OFF_ZN;
  float* out_ztex   = out + OFF_ZTEX;
  float* out_rw     = out + OFF_RW;
  float* out_zgeo   = out + OFF_ZGEO;
  float* out_idx    = out + OFF_IDX;
  float* out_znall  = out + OFF_ZNALL;

#pragma unroll
  for (int j = 0; j < 16; ++j) {
    int e = lane * 16 + j;
    zqbsh[w][e] = 0.0f;
    znbsh[w][e] = 0.0f;
  }

  v16bf ah[8];
#pragma unroll
  for (int ks = 0; ks < 8; ++ks) ah[ks] = load_a_bf16(h, HID, gr0, ks * 32);

  // ---- router scores = (h @ Mq^T + bq) / 16 ----
  v8f sc;
  {
    float bqc = bq[col];
#pragma unroll
    for (int r = 0; r < 8; ++r) sc[r] = bqc;
#pragma unroll
    for (int ks = 0; ks < 8; ++ks) {
      v16bf bf = load_b_bf16(Mqh, HID, 0, ks * 32);
      sc = WMMA_BF16(ah[ks], bf, sc);
    }
  }
#pragma unroll
  for (int r = 0; r < 8; ++r) {
    float s = sc[r] * 0.0625f;  // / sqrt(HID)
    float mx = s;
    int mi = col;
#pragma unroll
    for (int msk = 1; msk < 16; msk <<= 1) {
      float ov = __shfl_xor(mx, msk, 32);
      int oi = __shfl_xor(mi, msk, 32);
      if (ov > mx || (ov == mx && oi < mi)) { mx = ov; mi = oi; }
    }
    float e = __expf(s - mx);
    float sum = e;
#pragma unroll
    for (int msk = 1; msk < 16; msk <<= 1) sum += __shfl_xor(sum, msk, 32);
    float rw = e / sum;
    rwsh[w][(mbase + r) * 16 + col] = rw;
    if (col < NC) out_rw[(gr0 + mbase + r) * NC + col] = rw;
    if (col == 0) {
      kchsh[w][mbase + r] = mi;
      out_Kchart[gr0 + mbase + r] = (float)mi;
    }
  }

  // ---- v = h @ Wv^T + bv ----
#pragma unroll
  for (int ct = 0; ct < 2; ++ct) {
    v8f vt;
    float bvc = bv[ct * 16 + col];
#pragma unroll
    for (int r = 0; r < 8; ++r) vt[r] = bvc;
#pragma unroll
    for (int ks = 0; ks < 8; ++ks) {
      v16bf bf = load_b_bf16(Wvh, HID, ct * 16, ks * 32);
      vt = WMMA_BF16(ah[ks], bf, vt);
    }
#pragma unroll
    for (int r = 0; r < 8; ++r)
      vsh[w][(mbase + r) * LAT + ct * 16 + col] = vt[r];
  }
  __syncthreads();

  v16bf av = load_a_f32(&vsh[w][0], LAT, 0, 0);
  v16bf bws1 = load_b_f32(Ws1, LAT, 0, 0);   // 16 hidden cols, K=32
  v16bf bws2[2];
  bws2[0] = load_b_f32_pad(Ws2, 16, 0, 16);  // K padded 16 -> 32
  bws2[1] = load_b_f32_pad(Ws2, 16, 16, 16);

  float lsum = 0.0f;

  for (int n = 0; n < NC; ++n) {
    const float* cbn = cb + n * CODES * LAT;
    const __bf16* cbhn = cbh + n * CODES * LAT;

    // nearest code: argmin_c ||c||^2 - 2 v·c  (||v||^2 constant per row)
    float bvv[8];
    int bii[8];
#pragma unroll
    for (int r = 0; r < 8; ++r) { bvv[r] = 3.0e38f; bii[r] = 0; }
#pragma unroll 4
    for (int ct = 0; ct < 16; ++ct) {
      int c0 = ct * 16;
      v16bf bf = load_b_bf16(cbhn, LAT, c0, 0);
      v8f cr;
#pragma unroll
      for (int r = 0; r < 8; ++r) cr[r] = 0.0f;
      cr = WMMA_BF16(av, bf, cr);
      int c = c0 + col;
      float cs = csq[n * CODES + c];
#pragma unroll
      for (int r = 0; r < 8; ++r) {
        float dist = cs - 2.0f * cr[r];
        if (dist < bvv[r]) { bvv[r] = dist; bii[r] = c; }
      }
    }
#pragma unroll
    for (int r = 0; r < 8; ++r) {
#pragma unroll
      for (int msk = 1; msk < 16; msk <<= 1) {
        float ov = __shfl_xor(bvv[r], msk, 32);
        int oi = __shfl_xor(bii[r], msk, 32);
        if (ov < bvv[r] || (ov == bvv[r] && oi < bii[r])) { bvv[r] = ov; bii[r] = oi; }
      }
    }
    if (col == 0) {
#pragma unroll
      for (int r = 0; r < 8; ++r) {
        idxsh[w][(mbase + r) * NC + n] = bii[r];
        out_idx[(gr0 + mbase + r) * NC + n] = (float)bii[r];
      }
    }
    __syncthreads();

    // delta = v - z_q ; weighted loss ; z_q blend (z_q gathered in f32)
#pragma unroll
    for (int j = 0; j < 16; ++j) {
      int e = lane * 16 + j;
      int row = e >> 5, d = e & 31;
      float vv = vsh[w][e];
      int ci = idxsh[w][row * NC + n];
      float zq = cbn[ci * LAT + d];
      float dd = vv - zq;
      deltash[w][e] = dd;
      float wgt = rwsh[w][row * 16 + n];
      lsum += dd * dd * wgt;
      zqbsh[w][e] += zq * wgt;
    }
    __syncthreads();

    // chart MLP: t = gelu(delta @ Ws1^T + bs1)
    v16bf ad = load_a_f32(&deltash[w][0], LAT, 0, 0);
    v8f t;
    {
      float b0 = bs1[col];
#pragma unroll
      for (int r = 0; r < 8; ++r) t[r] = b0;
    }
    t = WMMA_BF16(ad, bws1, t);
#pragma unroll
    for (int r = 0; r < 8; ++r) tsh[w][(mbase + r) * 16 + col] = gelu_f(t[r]);
    __syncthreads();

    // z_chart = t @ Ws2^T + bs2  (K=16, zero-padded to 32)
    v16bf at = load_a_f32_pad(&tsh[w][0], 16, 0, 16);
#pragma unroll
    for (int ct = 0; ct < 2; ++ct) {
      v8f z;
      float b2c = bs2[ct * 16 + col];
#pragma unroll
      for (int r = 0; r < 8; ++r) z[r] = b2c;
      z = WMMA_BF16(at, bws2[ct], z);
      int cc = ct * 16 + col;
#pragma unroll
      for (int r = 0; r < 8; ++r) {
        int row = mbase + r;
        float wgt = rwsh[w][row * 16 + n];
        out_znall[((gr0 + row) * NC + n) * LAT + cc] = z[r];
        znbsh[w][row * LAT + cc] += z[r] * wgt;
      }
    }
    __syncthreads();
  }

  // ---- final blended outputs ----
#pragma unroll
  for (int j = 0; j < 16; ++j) {
    int e = lane * 16 + j;
    int row = e >> 5, d = e & 31;
    int gr = gr0 + row;
    float vv = vsh[w][e], zq = zqbsh[w][e], zn = znbsh[w][e];
    out_zn[gr * LAT + d] = zn;
    out_ztex[gr * LAT + d] = (vv - zq) - zn;  // delta_blended - z_n
    out_zgeo[gr * LAT + d] = zq + zn;         // z_q_st + z_n
  }
  if (lane < 16) {
    int row = lane;
    int kc = kchsh[w][row];
    out_Kcode[gr0 + row] = (float)idxsh[w][row * NC + kc];
  }

  // deterministic loss partials (no float atomics -> replay-stable)
#pragma unroll
  for (int msk = 1; msk < 32; msk <<= 1) lsum += __shfl_xor(lsum, msk, 32);
  if (lane == 0) redsh[w] = lsum;
  __syncthreads();
  if (tid == 0) part[blockIdx.x] = redsh[0] + redsh[1] + redsh[2] + redsh[3];
}

// ---------------------------------------------------------------------------
// K4: reduce 512 partials; forward codebook/commitment terms are identical,
// so vq_loss = 1.25 * sum / (B*LAT).
// ---------------------------------------------------------------------------
__global__ void k_fin(const float* __restrict__ part, float* __restrict__ out) {
  __shared__ float sh[256];
  int tid = threadIdx.x;
  sh[tid] = part[tid] + part[tid + 256];
  __syncthreads();
  for (int off = 128; off > 0; off >>= 1) {
    if (tid < off) sh[tid] += sh[tid + off];
    __syncthreads();
  }
  if (tid == 0) out[OFF_VQ] = sh[0] * (1.25f / (32768.0f * 32.0f));
}

// ---------------------------------------------------------------------------
extern "C" void kernel_launch(void* const* d_in, const int* in_sizes, int n_in,
                              void* d_out, int out_size, void* d_ws, size_t ws_size,
                              hipStream_t stream) {
  const float* x   = (const float*)d_in[0];
  const float* W1  = (const float*)d_in[1];
  const float* b1  = (const float*)d_in[2];
  const float* W2  = (const float*)d_in[3];
  const float* b2  = (const float*)d_in[4];
  const float* Wk  = (const float*)d_in[5];
  const float* bk  = (const float*)d_in[6];
  const float* cq  = (const float*)d_in[7];
  const float* Wv  = (const float*)d_in[8];
  const float* bv  = (const float*)d_in[9];
  const float* cb  = (const float*)d_in[10];
  const float* Ws1 = (const float*)d_in[11];
  const float* bs1 = (const float*)d_in[12];
  const float* Ws2 = (const float*)d_in[13];
  const float* bs2 = (const float*)d_in[14];

  char* ws = (char*)d_ws;
  __bf16* h1  = (__bf16*)(ws);                          // 16 MB
  __bf16* h   = (__bf16*)(ws + (size_t)16777216);       // 16 MB
  char* wb    = ws + (size_t)33554432;
  __bf16* W1h = (__bf16*)(wb);                          // 32 KB
  __bf16* W2h = (__bf16*)(wb + 32768);                  // 128 KB
  __bf16* Mqh = (__bf16*)(wb + 163840);                 // 8 KB (16x256, padded)
  __bf16* Wvh = (__bf16*)(wb + 172032);                 // 16 KB
  __bf16* cbh = (__bf16*)(wb + 188416);                 // 128 KB
  float*  bq  = (float*)(wb + 319488);                  // 16 f32
  float*  csq = (float*)(wb + 319552);                  // 8*256 f32
  float*  part = (float*)(wb + 327744);                 // 512 f32

  k0_precompute<<<1, 256, 0, stream>>>(Wk, bk, cq, cb, W1, W2, Wv,
                                       W1h, W2h, Mqh, Wvh, cbh, bq, csq);
  k1_gemm<<<BDIM / 64, 256, 0, stream>>>(x, W1h, b1, h1);
  k2_gemm<<<BDIM / 64, 256, 0, stream>>>(h1, W2h, b2, h);
  k3_fused<<<BDIM / 64, 128, 0, stream>>>(h, Mqh, bq, csq, Wvh, bv, cb, cbh,
                                          Ws1, bs1, Ws2, bs2, (float*)d_out, part);
  k_fin<<<1, 256, 0, stream>>>(part, (float*)d_out);
}
